// SimpleBlock_73778948211298
// MI455X (gfx1250) — matrix operand order repaired and checked by reading, not compiled, not run on hardware
//
#include <hip/hip_runtime.h>
#include <hip/hip_bf16.h>
#include <math.h>

// ---------------- problem constants (match reference) ----------------
#define NQ    65536
#define MS    65536
#define HN    32
#define INF_  64
#define OUTF  64
#define KP    15
#define KPAD  16           // k padded to 16 (row 15 is exactly zero)
#define KI    1024         // INF_ * KPAD, stage-B K dimension
#define NSTEP 32           // KI / 32 WMMA K-steps
#define EXTENT 0.048f      // 0.1 * 1.2 / 2.5
#define BN_EPS 1e-5f

typedef __attribute__((ext_vector_type(8)))  _Float16 v8h;
typedef __attribute__((ext_vector_type(16))) _Float16 v16h;
typedef __attribute__((ext_vector_type(8)))  float    v8f;

#define CAT16(lo, hi) __builtin_shufflevector((lo), (hi), 0,1,2,3,4,5,6,7,8,9,10,11,12,13,14,15)

// ---------------------------------------------------------------------
// Prep: zero BN accumulators; convert weights (K,IN,OUT) fp32 -> f16,
// pre-swizzled into the exact per-lane WMMA B-operand order so the main
// kernel's B fragment is one contiguous 32B load:
//   wB[ ((t*NSTEP + s)*32 + lane)*16 + j ]  with
//   col o = 16*t + (lane&15),  row r = 32*s + (lane>>4)*16 + j,
//   r = i*16 + k  ->  weights[k*4096 + i*64 + o]   (k==15 rows are zero)
// ---------------------------------------------------------------------
__global__ void kp_prep(const float* __restrict__ weights,
                        _Float16* __restrict__ wB,
                        float* __restrict__ gsum, float* __restrict__ gsq) {
  int id = blockIdx.x * blockDim.x + threadIdx.x;
  if (id < 64)        gsum[id] = 0.f;
  else if (id < 128)  gsq[id - 64] = 0.f;
  if (id >= 4 * NSTEP * 32) return;
  int lane  = id & 31;
  int s     = (id >> 5) % NSTEP;
  int t     = id / (NSTEP * 32);
  int o     = t * 16 + (lane & 15);
  int rbase = 32 * s + ((lane >> 4) << 4);
  v8h o0, o1;
#pragma unroll
  for (int j = 0; j < 8; ++j) {
    int r0 = rbase + j;
    int r1 = rbase + 8 + j;
    int k0 = r0 & 15, i0 = r0 >> 4;
    int k1 = r1 & 15, i1 = r1 >> 4;
    o0[j] = (k0 < KP) ? (_Float16)weights[k0 * 4096 + i0 * 64 + o] : (_Float16)0.f;
    o1[j] = (k1 < KP) ? (_Float16)weights[k1 * 4096 + i1 * 64 + o] : (_Float16)0.f;
  }
  *(v8h*)&wB[(size_t)id * 16 + 0] = o0;
  *(v8h*)&wB[(size_t)id * 16 + 8] = o1;
}

// ---------------------------------------------------------------------
// Main: one 128-thread block (4 wave32) per 16-query tile.
// Phase 1 (per wave, 4 queries each): gather s_pts/features, compute
//   kernel-point influences, stage wT (A) and feat (B) in LDS in WMMA
//   layouts, run 4x v_wmma_f32_16x16x32_f16 -> Wgt[16 x 1024] f16 in
//   LDS (i-major, k padded to 16 -> contiguous b128 writeback).
// Phase 2 (wave w = N-tile w): 32 K-step WMMAs against pre-swizzled
//   weights -> out_raw[16 x 64]; store to d_out (pre-BN) and accumulate
//   per-channel sum / sumsq for BatchNorm.
// ---------------------------------------------------------------------
__global__ __launch_bounds__(128) void kp_main(
    const float* __restrict__ x, const float* __restrict__ q_pts,
    const float* __restrict__ s_pts, const int* __restrict__ inds,
    const float* __restrict__ kpts, const _Float16* __restrict__ wB,
    float* __restrict__ out, float* __restrict__ gsum, float* __restrict__ gsq) {
  __shared__ __align__(16) _Float16 lds_feat[4][INF_ * HN]; // [wave][i*32 + h]
  __shared__ __align__(16) _Float16 lds_w[4][HN * 16];      // [wave][h*16 + k]
  __shared__ __align__(16) _Float16 lds_wgt[16 * KI];       // [q_local*1024 + i*16 + k]
  __shared__ float lds_kp[KP * 3];
  __shared__ float lds_sum[OUTF];
  __shared__ float lds_sq[OUTF];

  const int tid  = threadIdx.x;
  const int lane = tid & 31;
  const int wave = tid >> 5;
  const int q0   = blockIdx.x * 16;

  if (tid < KP * 3) lds_kp[tid] = kpts[tid];
  if (tid < OUTF) { lds_sum[tid] = 0.f; lds_sq[tid] = 0.f; }
  __syncthreads();

  // ------------------------- phase 1 -------------------------
  for (int qq = 0; qq < 4; ++qq) {
    const int q_local = wave * 4 + qq;
    const int q = q0 + q_local;

    // neighbor h = lane. Clamp index, load unconditionally (always safe),
    // then apply shadow-point semantics with selects (no exec branching).
    int ind = inds[q * HN + lane];
    const bool valid = ((unsigned)ind) < MS;
    const int indc = valid ? ind : 0;
    const float fs = valid ? 1.f : 0.f;
    const float qx = q_pts[(size_t)q * 3 + 0];
    const float qy = q_pts[(size_t)q * 3 + 1];
    const float qz = q_pts[(size_t)q * 3 + 2];
    const float* sp = s_pts + (size_t)indc * 3;
    float sx = sp[0], sy = sp[1], sz = sp[2];
    float dx = valid ? (sx - qx) : 1e6f;
    float dy = valid ? (sy - qy) : 1e6f;
    float dz = valid ? (sz - qz) : 1e6f;
    v8h wlo, whi;
#pragma unroll
    for (int k = 0; k < 16; ++k) {
      float wv = 0.f;
      if (k < KP) {
        float ax = dx - lds_kp[k * 3 + 0];
        float ay = dy - lds_kp[k * 3 + 1];
        float az = dz - lds_kp[k * 3 + 2];
        float d = sqrtf(ax * ax + ay * ay + az * az);
        wv = fmaxf(1.f - d * (1.f / EXTENT), 0.f);
      }
      if (k < 8) wlo[k] = (_Float16)wv; else whi[k - 8] = (_Float16)wv;
    }
    *(v8h*)&lds_w[wave][lane * 16 + 0] = wlo;
    *(v8h*)&lds_w[wave][lane * 16 + 8] = whi;

    // gather this neighbor's 64 features, store column-major f16 [i][h]
    const float4* xr = (const float4*)(x + (size_t)indc * INF_);
#pragma unroll
    for (int i4 = 0; i4 < 16; ++i4) {
      float4 f = xr[i4];
      int i = i4 * 4;
      lds_feat[wave][(i + 0) * HN + lane] = (_Float16)(f.x * fs);
      lds_feat[wave][(i + 1) * HN + lane] = (_Float16)(f.y * fs);
      lds_feat[wave][(i + 2) * HN + lane] = (_Float16)(f.z * fs);
      lds_feat[wave][(i + 3) * HN + lane] = (_Float16)(f.w * fs);
    }
    __syncthreads();  // cross-lane LDS visibility before WMMA fragment loads

    // A fragment: row M = k = lane&15; halves j<8 -> h=base+j, j>=8 -> h=base+16+(j-8)
    const int kcol = lane & 15;
    const int hbase = (lane < 16) ? 0 : 8;
    v16h a;
#pragma unroll
    for (int j = 0; j < 8; ++j) {
      a[j]     = lds_w[wave][(hbase + j) * 16 + kcol];
      a[8 + j] = lds_w[wave][(hbase + 16 + j) * 16 + kcol];
    }

    const int krow0 = (lane >> 4) * 8;  // D rows held by this lane (= k)
#pragma unroll
    for (int t = 0; t < 4; ++t) {
      // B fragment: col i = 16t + (lane&15), rows h: j<8 -> (lane>>4)*8 + j, j>=8 -> 16 + (lane>>4)*8 + (j-8)
      const int icol = t * 16 + (lane & 15);
      const _Float16* bp = &lds_feat[wave][icol * HN + ((lane >> 4) << 3)];
      v16h b = CAT16(*(const v8h*)bp, *(const v8h*)(bp + 16));
      v8f c = {};
      c = __builtin_amdgcn_wmma_f32_16x16x32_f16(false, a, false, b,
                                                 (short)0, c, false, false);
      // i-major Wgt: 8 consecutive k for fixed i -> one contiguous b128 store
      v8h cw;
#pragma unroll
      for (int v = 0; v < 8; ++v) cw[v] = (_Float16)c[v];
      *(v8h*)&lds_wgt[q_local * KI + icol * KPAD + krow0] = cw;
    }
  }
  __syncthreads();

  // ------------------------- phase 2 -------------------------
  {
    const int t = wave;                  // N-tile (output cols 16t..16t+15)
    const int qrow = lane & 15;          // A row = query-in-tile
    const int off = (lane < 16) ? 0 : 16; // A lane-group K offset
    v8f acc = {};
    const _Float16* wBb = wB + ((size_t)(t * NSTEP) * 32 + lane) * 16;
#pragma unroll 8
    for (int s = 0; s < NSTEP; ++s) {
      const _Float16* ap = &lds_wgt[qrow * KI + 32 * s + off];
      v16h a = CAT16(*(const v8h*)ap, *(const v8h*)(ap + 8));
      const _Float16* bp = wBb + (size_t)s * 32 * 16;
      v16h b = CAT16(*(const v8h*)bp, *(const v8h*)(bp + 8));
      acc = __builtin_amdgcn_wmma_f32_16x16x32_f16(false, a, false, b,
                                                   (short)0, acc, false, false);
    }
    const int o = t * 16 + (lane & 15);
    const int qr0 = (lane >> 4) * 8;
    float s1 = 0.f, s2 = 0.f;
#pragma unroll
    for (int v = 0; v < 8; ++v) {
      float val = acc[v];
      out[(size_t)(q0 + qr0 + v) * OUTF + o] = val;  // pre-BN activations
      s1 += val;
      s2 += val * val;
    }
    atomicAdd(&lds_sum[o], s1);  // ds_add_f32 (2 lanes per column)
    atomicAdd(&lds_sq[o], s2);
  }
  __syncthreads();
  if (tid < OUTF)           atomicAdd(&gsum[tid], lds_sum[tid]);
  else if (tid < 2 * OUTF)  atomicAdd(&gsq[tid - OUTF], lds_sq[tid - OUTF]);
}

// ---------------------------------------------------------------------
// Finalize: training-mode BatchNorm over points + LeakyReLU(0.1), in
// place, float4-vectorized (pure bandwidth: 16MB read + 16MB write).
// ---------------------------------------------------------------------
__global__ void kp_bn(float* __restrict__ out, const float* __restrict__ gamma,
                      const float* __restrict__ beta,
                      const float* __restrict__ gsum, const float* __restrict__ gsq) {
  int idx = blockIdx.x * blockDim.x + threadIdx.x;   // one float4 per thread
  if (idx >= NQ * OUTF / 4) return;
  int o0 = (idx * 4) & 63;
  float4* p = (float4*)out + idx;
  float4 v = *p;
  float r[4] = {v.x, v.y, v.z, v.w};
#pragma unroll
  for (int c = 0; c < 4; ++c) {
    int o = o0 + c;
    float mean = gsum[o] * (1.f / NQ);
    float var = gsq[o] * (1.f / NQ) - mean * mean;
    float inv = rsqrtf(var + BN_EPS);
    float y = (r[c] - mean) * inv * gamma[o] + beta[o];
    r[c] = (y >= 0.f) ? y : 0.1f * y;
  }
  *p = make_float4(r[0], r[1], r[2], r[3]);
}

// ---------------------------------------------------------------------
extern "C" void kernel_launch(void* const* d_in, const int* in_sizes, int n_in,
                              void* d_out, int out_size, void* d_ws, size_t ws_size,
                              hipStream_t stream) {
  const float* x       = (const float*)d_in[0];
  const float* q_pts   = (const float*)d_in[1];
  const float* s_pts   = (const float*)d_in[2];
  const int*   inds    = (const int*)d_in[3];
  const float* kpts    = (const float*)d_in[4];
  const float* weights = (const float*)d_in[5];
  const float* gamma   = (const float*)d_in[6];
  const float* beta    = (const float*)d_in[7];
  float* out = (float*)d_out;

  char* ws = (char*)d_ws;
  _Float16* wB = (_Float16*)ws;                     // 131072 B: swizzled f16 weights
  float* gsum  = (float*)(ws + 131072);             // 64 f32
  float* gsq   = (float*)(ws + 131072 + 256);       // 64 f32

  kp_prep<<<16, 256, 0, stream>>>(weights, wB, gsum, gsq);          // 4096 threads
  kp_main<<<NQ / 16, 128, 0, stream>>>(x, q_pts, s_pts, inds, kpts,
                                       wB, out, gsum, gsq);
  kp_bn<<<(NQ * OUTF / 4 + 255) / 256, 256, 0, stream>>>(out, gamma, beta, gsum, gsq);
}
